// NeuronMamba2Layer_68899865362583
// MI455X (gfx1250) — matrix operand order repaired and checked by reading, not compiled, not run on hardware
//
#include <hip/hip_runtime.h>
#include <hip/hip_bf16.h>
#include <math.h>

// ---------------------------------------------------------------------------
// Mamba2 layer for MI455X (gfx1250, wave32, WMMA + TDM).
//   proj GEMM (bf16 WMMA, B tiles staged in LDS via tensor_load_to_lds) ->
//   conv+SiLU -> sequential SSM scan (regs) -> gated RMSNorm (bf16) ->
//   out GEMM (bf16 WMMA + TDM)
// Workspace layout (bytes, ~362 MB total):
//   proj      : f32 [4096][8512]   @ 0
//   xBC_act   : f32 [4096][4352]   @ 139,460,608
//   y         : f32 [4096][4096]   @ 210,763,776
//   hs_bf16   :     [4096][2048]   @ 277,872,640
//   Win_bf16  :     [8512][2048]   @ 294,649,856
//   Wout_bf16 :     [2048][4096]   @ 329,515,008
//   yn_bf16   :     [4096][4096]   @ 346,292,224
// ---------------------------------------------------------------------------

typedef __attribute__((ext_vector_type(16))) __bf16 v16bf;
typedef __attribute__((ext_vector_type(8)))  __bf16 v8bf;
typedef __attribute__((ext_vector_type(8)))  float  v8f;
typedef __attribute__((ext_vector_type(4)))  float  v4f;
typedef __attribute__((ext_vector_type(4)))  unsigned int u32x4;
typedef __attribute__((ext_vector_type(8)))  unsigned int u32x8;

#define HID_D    2048
#define I_DIM    4096
#define H_HEADS  64
#define P_DIM    64
#define N_STATE  128
#define K_CONV   4
#define GTS      128
#define CONV_DIM 4352            // I + 2*GTS
#define PROJ_DIM 8512            // I + CONV_DIM + H
#define BATCH    2
#define SEQ      2048
#define M_TOK    4096            // BATCH*SEQ

// ----------------------------- f32 -> bf16 ---------------------------------
__global__ void cvt_f32_to_bf16(const float* __restrict__ src,
                                __bf16* __restrict__ dst, int n4) {
  int i = blockIdx.x * blockDim.x + threadIdx.x;
  if (i >= n4) return;
  v4f v = *(const v4f*)(src + (size_t)i * 4);
  __bf16* d = dst + (size_t)i * 4;
  d[0] = (__bf16)v.x; d[1] = (__bf16)v.y; d[2] = (__bf16)v.z; d[3] = (__bf16)v.w;
}

// ---------------- TDM issue: DMA a 64(row) x 32(K) bf16 tile to LDS --------
// D# group0: count=1 | lds_addr | global_addr | type=2  (2D tensor -> 2 groups)
__device__ __forceinline__ void tdm_load_b_tile(unsigned lds_off,
                                                const __bf16* gptr,
                                                u32x8 g1) {
  unsigned long long ga = (unsigned long long)(uintptr_t)gptr;
  u32x4 g0;
  g0.x = 1u;                                    // count = 1 valid descriptor
  g0.y = lds_off;                               // LDS byte address
  g0.z = (unsigned)ga;                          // global_addr[31:0]
  g0.w = (unsigned)(ga >> 32) | 0x80000000u;    // global_addr[56:32] | type=2
  asm volatile("tensor_load_to_lds %0, %1" :: "s"(g0), "s"(g1) : "memory");
}

// ------------------------ bf16 WMMA GEMM:  C = A * B^T ---------------------
// A: [M][K] bf16 row-major, B: [N][K] bf16 row-major, C: [M][Ndim] f32.
// Block = 8 waves -> 128(M) x 64(N) tile. B tile (64x32) staged in LDS by the
// Tensor Data Mover (double-buffered, TENSORcnt), shared by all 8 waves.
// Each wave: 16(M) x 64(N) = 4 WMMAs per K-step sharing one A fragment.
__global__ void gemm_bf16_wmma(const __bf16* __restrict__ A,
                               const __bf16* __restrict__ B,
                               float* __restrict__ C,
                               int Ntiles64, int Kdim, int Ndim) {
  __shared__ __bf16 Bs[2][64 * 32];             // 2 x 4 KB, [col][k] contiguous
  const int tid  = threadIdx.x;
  const int lane = tid & 31;
  const int wv   = tid >> 5;
  const int mb   = blockIdx.x / Ntiles64;
  const int nt   = blockIdx.x % Ntiles64;
  const int m0   = mb * 128 + wv * 16;
  const int n0   = nt * 64;
  const int half = lane >> 4;                   // 0: lanes 0-15, 1: lanes 16-31
  const int lr   = lane & 15;

  const __bf16* arow  = A + (size_t)(m0 + lr) * Kdim;
  const __bf16* btile = B + (size_t)n0 * Kdim;  // tile row base (col n0, k 0)

  // D# group1: data_size=2B, tensor_dim0=K, tensor_dim1=N rows,
  // tile_dim0=32 (K), tile_dim1=64 (rows), tensor_dim0_stride=K.
  const unsigned kd = (unsigned)Kdim, nd = (unsigned)Ndim;
  u32x8 g1;
  g1[0] = 0x00010000u;                          // data_size = 2 bytes
  g1[1] = (kd & 0xFFFFu) << 16;                 // tensor_dim0[15:0]  @ bits 63:48
  g1[2] = (kd >> 16) | ((nd & 0xFFFFu) << 16);  // dim0[31:16] | dim1[15:0]
  g1[3] = (nd >> 16) | (32u << 16);             // dim1[31:16] | tile_dim0 = 32
  g1[4] = 64u;                                  // tile_dim1 = 64, tile_dim2 = 0
  g1[5] = kd;                                   // tensor_dim0_stride[31:0]
  g1[6] = 0u;                                   // stride hi / dim1_stride lo
  g1[7] = 0u;

  const unsigned lds_base = (unsigned)(uintptr_t)(&Bs[0][0]);

  v8f acc0 = {}; v8f acc1 = {}; v8f acc2 = {}; v8f acc3 = {};
  const int nK = Kdim >> 5;

  if (wv == 0) tdm_load_b_tile(lds_base, btile, g1);      // prime buffer 0

  for (int kt = 0; kt < nK; kt++) {
    const int k = kt << 5;
    if (wv == 0) {
      if (kt + 1 < nK) {
        // stage next tile while this one is consumed; TDM ops are in-order
        tdm_load_b_tile(lds_base + (unsigned)(((kt + 1) & 1) * 4096),
                        btile + (k + 32), g1);
        __builtin_amdgcn_s_wait_tensorcnt(1);             // tile kt landed
      } else {
        __builtin_amdgcn_s_wait_tensorcnt(0);
      }
    }
    __syncthreads();                                      // LDS tile visible

    // A fragment (direct from global): row m0+lr, K chunks {kA, kA+8}
    const int kA = k + half * 16;
    __builtin_prefetch(arow + kA + 256, 0, 1);            // global_prefetch_b8
    v8bf alo = *(const v8bf*)(arow + kA);
    v8bf ahi = *(const v8bf*)(arow + kA + 8);
    v16bf af = __builtin_shufflevector(alo, ahi, 0,1,2,3,4,5,6,7,
                                                 8,9,10,11,12,13,14,15);

    // B fragments from LDS: lane-half picks 16 contiguous K values per column
    const __bf16* bbuf = &Bs[kt & 1][half * 16];
    {
      const __bf16* bp = bbuf + (0  + lr) * 32;
      v8bf b0 = *(const v8bf*)(bp);
      v8bf b1 = *(const v8bf*)(bp + 8);
      v16bf bf = __builtin_shufflevector(b0, b1, 0,1,2,3,4,5,6,7,
                                                 8,9,10,11,12,13,14,15);
      acc0 = __builtin_amdgcn_wmma_f32_16x16x32_bf16(false, af, false, bf,
                                                     (short)0, acc0, false, false);
    }
    {
      const __bf16* bp = bbuf + (16 + lr) * 32;
      v8bf b0 = *(const v8bf*)(bp);
      v8bf b1 = *(const v8bf*)(bp + 8);
      v16bf bf = __builtin_shufflevector(b0, b1, 0,1,2,3,4,5,6,7,
                                                 8,9,10,11,12,13,14,15);
      acc1 = __builtin_amdgcn_wmma_f32_16x16x32_bf16(false, af, false, bf,
                                                     (short)0, acc1, false, false);
    }
    {
      const __bf16* bp = bbuf + (32 + lr) * 32;
      v8bf b0 = *(const v8bf*)(bp);
      v8bf b1 = *(const v8bf*)(bp + 8);
      v16bf bf = __builtin_shufflevector(b0, b1, 0,1,2,3,4,5,6,7,
                                                 8,9,10,11,12,13,14,15);
      acc2 = __builtin_amdgcn_wmma_f32_16x16x32_bf16(false, af, false, bf,
                                                     (short)0, acc2, false, false);
    }
    {
      const __bf16* bp = bbuf + (48 + lr) * 32;
      v8bf b0 = *(const v8bf*)(bp);
      v8bf b1 = *(const v8bf*)(bp + 8);
      v16bf bf = __builtin_shufflevector(b0, b1, 0,1,2,3,4,5,6,7,
                                                 8,9,10,11,12,13,14,15);
      acc3 = __builtin_amdgcn_wmma_f32_16x16x32_bf16(false, af, false, bf,
                                                     (short)0, acc3, false, false);
    }
    __syncthreads();          // all waves done reading before buffer re-staged
  }

  // C/D layout: VGPR i, lanes 0-15 -> (M=i, N=lane); lanes 16-31 -> (M=8+i)
  #pragma unroll
  for (int i = 0; i < 8; i++) {
    const size_t row = (size_t)(m0 + 8 * half + i) * Ndim;
    C[row + n0 + 0  + lr] = acc0[i];
    C[row + n0 + 16 + lr] = acc1[i];
    C[row + n0 + 32 + lr] = acc2[i];
    C[row + n0 + 48 + lr] = acc3[i];
  }
}

// --------------------- depthwise causal conv (K=4) + SiLU ------------------
__global__ void conv_silu(const float* __restrict__ proj,
                          const float* __restrict__ cw,
                          const float* __restrict__ cb,
                          float* __restrict__ xact,
                          float* __restrict__ conv_state_out) {
  int idx = blockIdx.x * blockDim.x + threadIdx.x;
  if (idx >= BATCH * CONV_DIM) return;
  const int b = idx / CONV_DIM;
  const int c = idx % CONV_DIM;
  const float w0 = cw[c * K_CONV + 0], w1 = cw[c * K_CONV + 1];
  const float w2 = cw[c * K_CONV + 2], w3 = cw[c * K_CONV + 3];
  const float bias = cb[c];
  const float* src = proj + (size_t)b * SEQ * PROJ_DIM + I_DIM + c;
  float* dst = xact + (size_t)b * SEQ * CONV_DIM + c;
  float h0 = 0.f, h1 = 0.f, h2 = 0.f;   // xBC[t-3], xBC[t-2], xBC[t-1]
  for (int t = 0; t < SEQ; t++) {
    const float v = src[(size_t)t * PROJ_DIM];
    const float s = h0 * w0 + h1 * w1 + h2 * w2 + v * w3 + bias;
    dst[(size_t)t * CONV_DIM] = s / (1.f + __expf(-s));   // SiLU
    h0 = h1; h1 = h2; h2 = v;
  }
  // conv_state_new[b][c][k] = xBC[b][SEQ-3+k][c]  (pre-activation)
  float* cs = conv_state_out + ((size_t)b * CONV_DIM + c) * (K_CONV - 1);
  cs[0] = h0; cs[1] = h1; cs[2] = h2;
}

// --------------------------- sequential SSM scan ---------------------------
// One block per (b,h); 256 threads: lane owns (p = tid/4, 32 states of n).
__global__ void ssm_scan(const float* __restrict__ proj,
                         const float* __restrict__ xact,
                         const float* __restrict__ dt_bias,
                         const float* __restrict__ A_log,
                         const float* __restrict__ Dv,
                         float* __restrict__ y,
                         float* __restrict__ ssm_state_out) {
  const int b = blockIdx.x / H_HEADS;
  const int h = blockIdx.x % H_HEADS;
  const int tid = threadIdx.x;
  const int p = tid >> 2;
  const int q = tid & 3;
  const int nbase = q * 32;

  const float Ah = -__expf(A_log[h]);
  const float dbias = dt_bias[h];
  const float Dh = Dv[h];

  float st[32];
  #pragma unroll
  for (int i = 0; i < 32; i++) st[i] = 0.f;

  const float* dt_ptr = proj + (size_t)b * SEQ * PROJ_DIM + (I_DIM + CONV_DIM) + h;
  const float* xa = xact + (size_t)b * SEQ * CONV_DIM;

  for (int t = 0; t < SEQ; t++) {
    const float* row = xa + (size_t)t * CONV_DIM;
    float dtr = dt_ptr[(size_t)t * PROJ_DIM] + dbias;
    float sp  = (dtr > 20.f) ? dtr : log1pf(__expf(dtr));     // softplus
    float dtp = fminf(sp, 1000000.f);                          // clip to DT_MAX
    float dA  = __expf(dtp * Ah);
    float xp  = row[h * P_DIM + p];
    float xdt = xp * dtp;
    float yp = 0.f;
    #pragma unroll
    for (int i = 0; i < 32; i++) {
      const float Bn = row[I_DIM + nbase + i];
      const float Cn = row[I_DIM + GTS + nbase + i];
      st[i] = fmaf(st[i], dA, xdt * Bn);
      yp = fmaf(st[i], Cn, yp);
    }
    yp += __shfl_xor(yp, 1, 32);
    yp += __shfl_xor(yp, 2, 32);
    if (q == 0)
      y[((size_t)b * SEQ + t) * I_DIM + h * P_DIM + p] = yp + Dh * xp;
  }
  float* so = ssm_state_out + (((size_t)b * H_HEADS + h) * P_DIM + p) * N_STATE + nbase;
  #pragma unroll
  for (int i = 0; i < 32; i++) so[i] = st[i];
}

// --------------------- gate*SiLU + RMSNorm -> bf16 -------------------------
__global__ void gated_rmsnorm(const float* __restrict__ y,
                              const float* __restrict__ proj,
                              const float* __restrict__ nw,
                              __bf16* __restrict__ yn) {
  const int row = blockIdx.x;                  // token index 0..4095
  const float* yr = y + (size_t)row * I_DIM;
  const float* gr = proj + (size_t)row * PROJ_DIM;   // gate = proj[:, :I]
  float vals[16];
  float local = 0.f;
  #pragma unroll
  for (int j = 0; j < 16; j++) {
    const int i = threadIdx.x + j * 256;
    const float g = gr[i];
    const float yg = yr[i] * (g / (1.f + __expf(-g)));
    vals[j] = yg;
    local = fmaf(yg, yg, local);
  }
  #pragma unroll
  for (int off = 16; off > 0; off >>= 1) local += __shfl_xor(local, off, 32);
  __shared__ float wsum[8];
  if ((threadIdx.x & 31) == 0) wsum[threadIdx.x >> 5] = local;
  __syncthreads();
  float tot = 0.f;
  #pragma unroll
  for (int w = 0; w < 8; w++) tot += wsum[w];
  const float scale = rsqrtf(tot * (1.f / I_DIM) + 1e-6f);
  __bf16* outr = yn + (size_t)row * I_DIM;
  #pragma unroll
  for (int j = 0; j < 16; j++) {
    const int i = threadIdx.x + j * 256;
    outr[i] = (__bf16)(vals[j] * scale * nw[i]);
  }
}

// ---------------------------------------------------------------------------
extern "C" void kernel_launch(void* const* d_in, const int* in_sizes, int n_in,
                              void* d_out, int out_size, void* d_ws, size_t ws_size,
                              hipStream_t stream) {
  const float* hs        = (const float*)d_in[0];
  const float* W_in      = (const float*)d_in[1];
  const float* conv_w    = (const float*)d_in[2];
  const float* conv_b    = (const float*)d_in[3];
  const float* dt_bias   = (const float*)d_in[4];
  const float* A_log     = (const float*)d_in[5];
  const float* Dv        = (const float*)d_in[6];
  const float* norm_w    = (const float*)d_in[7];
  const float* W_out     = (const float*)d_in[8];

  // Output sections (return order): out, conv_state_new, ssm_state_new
  float* out        = (float*)d_out;                                   // 8,388,608
  float* conv_state = out + (size_t)M_TOK * HID_D;                     // 26,112
  float* ssm_state  = conv_state + (size_t)BATCH * CONV_DIM * (K_CONV - 1);

  // Workspace carve-up
  char* ws = (char*)d_ws;
  float*  proj    = (float*)(ws);
  float*  xact    = (float*)(ws + 139460608ull);
  float*  ybuf    = (float*)(ws + 210763776ull);
  __bf16* hs_bf   = (__bf16*)(ws + 277872640ull);
  __bf16* win_bf  = (__bf16*)(ws + 294649856ull);
  __bf16* wout_bf = (__bf16*)(ws + 329515008ull);
  __bf16* yn_bf   = (__bf16*)(ws + 346292224ull);

  // 1) one-time bf16 conversions of GEMM operands
  {
    int n4 = (M_TOK * HID_D) / 4;                 // hidden_states
    cvt_f32_to_bf16<<<(n4 + 255) / 256, 256, 0, stream>>>(hs, hs_bf, n4);
    n4 = (PROJ_DIM * HID_D) / 4;                  // W_in
    cvt_f32_to_bf16<<<(n4 + 255) / 256, 256, 0, stream>>>(W_in, win_bf, n4);
    n4 = (HID_D * I_DIM) / 4;                     // W_out
    cvt_f32_to_bf16<<<(n4 + 255) / 256, 256, 0, stream>>>(W_out, wout_bf, n4);
  }

  // 2) input projection: proj[4096][8512] = hs * W_in^T (bf16 WMMA + TDM)
  {
    const int Nt = PROJ_DIM / 64;                 // 133
    const int blocks = (M_TOK / 128) * Nt;        // 32 * 133 = 4256
    gemm_bf16_wmma<<<blocks, 256, 0, stream>>>(hs_bf, win_bf, proj,
                                               Nt, HID_D, PROJ_DIM);
  }

  // 3) depthwise conv + SiLU + conv_state
  {
    const int n = BATCH * CONV_DIM;               // 8704
    conv_silu<<<(n + 255) / 256, 256, 0, stream>>>(proj, conv_w, conv_b,
                                                   xact, conv_state);
  }

  // 4) sequential SSM scan (one workgroup per (b,h))
  ssm_scan<<<BATCH * H_HEADS, 256, 0, stream>>>(proj, xact, dt_bias, A_log,
                                                Dv, ybuf, ssm_state);

  // 5) gated SiLU + RMSNorm -> bf16 activations
  gated_rmsnorm<<<M_TOK, 256, 0, stream>>>(ybuf, proj, norm_w, yn_bf);

  // 6) output projection: out[4096][2048] = yn * W_out^T (bf16 WMMA + TDM)
  {
    const int Nt = HID_D / 64;                    // 32
    const int blocks = (M_TOK / 128) * Nt;        // 1024
    gemm_bf16_wmma<<<blocks, 256, 0, stream>>>(yn_bf, wout_bf, out,
                                               Nt, I_DIM, HID_D);
  }
}
